// ScienceIETagger_67319317397599
// MI455X (gfx1250) — compile-verified, matrix operand order checked by reading, not run on hardware
//
#include <hip/hip_runtime.h>
#include <math.h>

// ---------------------------------------------------------------------------
// ScienceIE multi-task CRF tagger for MI455X (gfx1250, wave32, WMMA)
//   B=64, T=512, H=1024, C=8, K(tasks)=3
//   d_out = [loss(1) | logits(3*B*T*C) | pred(3*B*T)]  (floats)
//   d_ws  = [backpointers 192*512*8 bytes | chain_loss 192 floats]
// ---------------------------------------------------------------------------

#define TT 512
#define BB 64
#define CC 8
#define HH 1024
#define NTASK 3
#define MROWS (BB * TT)          // 32768
#define NCHAIN (NTASK * BB)      // 192
#define BP_STRIDE ((size_t)TT * CC)   // 4096 bytes per chain

typedef __attribute__((ext_vector_type(16))) _Float16 v16h;
typedef __attribute__((ext_vector_type(8)))  float    v8f;

// ---------------------------------------------------------------------------
// Kernel 1: logits = einsum('bth,kch->kbtc') + b   via v_wmma_f32_16x16x32_f16
//   GEMM  M=32768, K=1024, N=24 (padded to 32).
//   One wave = one 16-row M-tile x BOTH 16-col N-tiles: the A fragment is
//   loaded from HBM exactly once and reused for two WMMAs (guaranteed
//   register-level reuse; encoding is streamed once -> ~134 MB total).
// ---------------------------------------------------------------------------
__global__ __launch_bounds__(256)
void scienceie_logits_wmma(const float* __restrict__ enc,    // [M, H]
                           const float* __restrict__ W,      // [3, 8, H]
                           const float* __restrict__ bias,   // [3, 8]
                           float* __restrict__ logits)       // [3, M, 8]
{
    const int lane  = threadIdx.x & 31;
    const int wave  = threadIdx.x >> 5;
    const int mtile = blockIdx.x * 8 + wave;   // 0..2047

    // A-fragment mapping (16-bit A 16x32): row = lane&15, K-half = lane>>4
    const int mrow = lane & 15;
    const int hi   = lane >> 4;
    const float* arow = enc + (size_t)(mtile * 16 + mrow) * HH;

    // B-fragment mapping (16-bit B 32x16): col = lane&15, same K pattern
    const int nloc = lane & 15;
    // N-tile 0: n = nloc (0..15), always a valid output column
    const int task0 = nloc >> 3;
    const int c0    = nloc & 7;
    const float* wcol0 = W + (size_t)(task0 * CC + c0) * HH;
    // N-tile 1: n = 16 + nloc, valid iff nloc < 8 (task 2); rest zero-padded
    const int nvalid1  = (nloc < 8);
    const float nmask1 = nvalid1 ? 1.0f : 0.0f;
    const float* wcol1 = W + (size_t)(2 * CC + (nvalid1 ? nloc : 0)) * HH;

    v8f acc0 = {};
    v8f acc1 = {};
    for (int kb = 0; kb < HH; kb += 32) {
        const float* ap  = arow  + kb + hi * 8;  // K = kb+hi*8..+7, +16..+23
        const float* bp0 = wcol0 + kb + hi * 8;
        const float* bp1 = wcol1 + kb + hi * 8;
        __builtin_prefetch(ap + 32, 0, 1);       // global_prefetch_b8

        float4 a0 = ((const float4*)ap)[0];
        float4 a1 = ((const float4*)ap)[1];
        float4 a2 = ((const float4*)(ap + 16))[0];
        float4 a3 = ((const float4*)(ap + 16))[1];
        float4 p0 = ((const float4*)bp0)[0];
        float4 p1 = ((const float4*)bp0)[1];
        float4 p2 = ((const float4*)(bp0 + 16))[0];
        float4 p3 = ((const float4*)(bp0 + 16))[1];
        float4 q0 = ((const float4*)bp1)[0];
        float4 q1 = ((const float4*)bp1)[1];
        float4 q2 = ((const float4*)(bp1 + 16))[0];
        float4 q3 = ((const float4*)(bp1 + 16))[1];

        v16h av, bv0, bv1;
        av[0]  = (_Float16)a0.x; av[1]  = (_Float16)a0.y;
        av[2]  = (_Float16)a0.z; av[3]  = (_Float16)a0.w;
        av[4]  = (_Float16)a1.x; av[5]  = (_Float16)a1.y;
        av[6]  = (_Float16)a1.z; av[7]  = (_Float16)a1.w;
        av[8]  = (_Float16)a2.x; av[9]  = (_Float16)a2.y;
        av[10] = (_Float16)a2.z; av[11] = (_Float16)a2.w;
        av[12] = (_Float16)a3.x; av[13] = (_Float16)a3.y;
        av[14] = (_Float16)a3.z; av[15] = (_Float16)a3.w;

        bv0[0]  = (_Float16)p0.x; bv0[1]  = (_Float16)p0.y;
        bv0[2]  = (_Float16)p0.z; bv0[3]  = (_Float16)p0.w;
        bv0[4]  = (_Float16)p1.x; bv0[5]  = (_Float16)p1.y;
        bv0[6]  = (_Float16)p1.z; bv0[7]  = (_Float16)p1.w;
        bv0[8]  = (_Float16)p2.x; bv0[9]  = (_Float16)p2.y;
        bv0[10] = (_Float16)p2.z; bv0[11] = (_Float16)p2.w;
        bv0[12] = (_Float16)p3.x; bv0[13] = (_Float16)p3.y;
        bv0[14] = (_Float16)p3.z; bv0[15] = (_Float16)p3.w;

        bv1[0]  = (_Float16)(q0.x * nmask1); bv1[1]  = (_Float16)(q0.y * nmask1);
        bv1[2]  = (_Float16)(q0.z * nmask1); bv1[3]  = (_Float16)(q0.w * nmask1);
        bv1[4]  = (_Float16)(q1.x * nmask1); bv1[5]  = (_Float16)(q1.y * nmask1);
        bv1[6]  = (_Float16)(q1.z * nmask1); bv1[7]  = (_Float16)(q1.w * nmask1);
        bv1[8]  = (_Float16)(q2.x * nmask1); bv1[9]  = (_Float16)(q2.y * nmask1);
        bv1[10] = (_Float16)(q2.z * nmask1); bv1[11] = (_Float16)(q2.w * nmask1);
        bv1[12] = (_Float16)(q3.x * nmask1); bv1[13] = (_Float16)(q3.y * nmask1);
        bv1[14] = (_Float16)(q3.z * nmask1); bv1[15] = (_Float16)(q3.w * nmask1);

        acc0 = __builtin_amdgcn_wmma_f32_16x16x32_f16(
            false, av, false, bv0, (short)0, acc0, false, false);
        acc1 = __builtin_amdgcn_wmma_f32_16x16x32_f16(
            false, av, false, bv1, (short)0, acc1, false, false);
    }

    // D layout: lane group hi, VGPR r -> M = r + hi*8 ; N = lane&15
    const int gmbase = mtile * 16 + hi * 8;
    {
        const float badd = bias[task0 * CC + c0];
        const size_t obase = (size_t)task0 * MROWS * CC + c0;
        #pragma unroll
        for (int r = 0; r < 8; ++r)
            logits[obase + (size_t)(gmbase + r) * CC] = acc0[r] + badd;
    }
    if (nvalid1) {
        const float badd = bias[2 * CC + nloc];
        const size_t obase = (size_t)2 * MROWS * CC + nloc;
        #pragma unroll
        for (int r = 0; r < 8; ++r)
            logits[obase + (size_t)(gmbase + r) * CC] = acc1[r] + badd;
    }
}

// ---------------------------------------------------------------------------
// Kernel 2: CRF forward (logZ), gold-path score, Viterbi + backtrace.
//   192 chains, 8 lanes/chain (lane = current state), 4 chains/wave32.
// ---------------------------------------------------------------------------
__global__ __launch_bounds__(32)
void scienceie_crf(const float* __restrict__ logits,   // [3, B*T, 8]
                   const int*   __restrict__ labels,   // [B, 3*T]
                   const float* __restrict__ start,    // [3, 8]
                   const float* __restrict__ endt,     // [3, 8]
                   const float* __restrict__ trans,    // [3, 8, 8]
                   float* __restrict__ pred,           // [3, B, T] (as float)
                   unsigned char* __restrict__ bpws,   // [192, T, 8]
                   float* __restrict__ chain_loss)     // [192]
{
    const int lane  = threadIdx.x;            // 0..31
    const int gb    = lane & 24;              // 8-lane group base
    const int c     = lane & 7;               // current-state index
    const int chain = blockIdx.x * 4 + (lane >> 3);   // 0..191
    const int k = chain >> 6;                 // task
    const int b = chain & 63;                 // batch

    const float* em  = logits + ((size_t)k * BB + b) * TT * CC;
    const int*  tags = labels + (size_t)b * (NTASK * TT) + k * TT;
    const float* tr  = trans + k * CC * CC;
    unsigned char* bp = bpws + (size_t)chain * BP_STRIDE;

    float trc[CC];                            // column c of trans[k]
    #pragma unroll
    for (int p = 0; p < CC; ++p) trc[p] = tr[p * CC + c];

    float alpha = start[k * CC + c] + em[c];  // forward log-alpha
    float vit   = alpha;                      // viterbi score

    for (int t = 1; t < TT; ++t) {
        const float e = em[t * CC + c];
        float sa[CC], sv[CC];
        #pragma unroll
        for (int p = 0; p < CC; ++p) {
            sa[p] = __shfl(alpha, gb | p) + trc[p];
            sv[p] = __shfl(vit,   gb | p) + trc[p];
        }
        float m = sa[0];
        #pragma unroll
        for (int p = 1; p < CC; ++p) m = fmaxf(m, sa[p]);
        float s = 0.0f;
        #pragma unroll
        for (int p = 0; p < CC; ++p) s += __expf(sa[p] - m);
        alpha = m + __logf(s) + e;

        float best = sv[0]; int barg = 0;
        #pragma unroll
        for (int p = 1; p < CC; ++p)
            if (sv[p] > best) { best = sv[p]; barg = p; }
        vit = best + e;
        bp[(size_t)(t - 1) * CC + c] = (unsigned char)barg;
    }

    // logZ and final-state argmax (redundant across lanes; trivially cheap)
    const float afin = alpha + endt[k * CC + c];
    const float vfin = vit   + endt[k * CC + c];
    float aarr[CC], varr[CC];
    #pragma unroll
    for (int p = 0; p < CC; ++p) {
        aarr[p] = __shfl(afin, gb | p);
        varr[p] = __shfl(vfin, gb | p);
    }
    float m = aarr[0];
    #pragma unroll
    for (int p = 1; p < CC; ++p) m = fmaxf(m, aarr[p]);
    float s = 0.0f;
    #pragma unroll
    for (int p = 0; p < CC; ++p) s += __expf(aarr[p] - m);
    const float logZ = m + __logf(s);

    float vbest = varr[0]; int last = 0;
    #pragma unroll
    for (int p = 1; p < CC; ++p)
        if (varr[p] > vbest) { vbest = varr[p]; last = p; }

    // gold-path score, strided over the 8 lanes of the group
    float part = 0.0f;
    for (int t = c; t < TT; t += CC) {
        const int tg = tags[t];
        part += em[t * CC + tg];
        if (t + 1 < TT) part += tr[tg * CC + tags[t + 1]];
    }
    #pragma unroll
    for (int off = 4; off > 0; off >>= 1) part += __shfl_xor(part, off);

    // order backpointer byte stores before the serial backtrace reads them
    asm volatile("s_wait_storecnt 0" ::: "memory");

    if (c == 0) {
        const float num = part + start[k * CC + tags[0]]
                               + endt[k * CC + tags[TT - 1]];
        chain_loss[chain] = logZ - num;       // loss contribution (-(num-logZ))

        float* pr = pred + ((size_t)k * BB + b) * TT;
        int cur = last;
        pr[TT - 1] = (float)cur;
        for (int t = TT - 1; t >= 1; --t) {
            cur = bp[(size_t)(t - 1) * CC + cur];
            pr[t - 1] = (float)cur;
        }
    }
}

// ---------------------------------------------------------------------------
// Kernel 3: deterministic scalar loss reduction (192 -> 1)
// ---------------------------------------------------------------------------
__global__ __launch_bounds__(32)
void scienceie_loss_reduce(const float* __restrict__ chain_loss,
                           float* __restrict__ loss)
{
    float s = 0.0f;
    for (int i = threadIdx.x; i < NCHAIN; i += 32) s += chain_loss[i];
    #pragma unroll
    for (int off = 16; off > 0; off >>= 1) s += __shfl_xor(s, off);
    if (threadIdx.x == 0) loss[0] = s;
}

// ---------------------------------------------------------------------------
extern "C" void kernel_launch(void* const* d_in, const int* in_sizes, int n_in,
                              void* d_out, int out_size, void* d_ws, size_t ws_size,
                              hipStream_t stream) {
    const float* enc    = (const float*)d_in[0];   // [B,T,H]
    const int*   labels = (const int*)  d_in[1];   // [B,3T]
    const float* W      = (const float*)d_in[2];   // [3,C,H]
    const float* bias   = (const float*)d_in[3];   // [3,C]
    const float* start  = (const float*)d_in[4];   // [3,C]
    const float* endt   = (const float*)d_in[5];   // [3,C]
    const float* trans  = (const float*)d_in[6];   // [3,C,C]

    float* out    = (float*)d_out;
    float* loss   = out;                               // [1]
    float* logits = out + 1;                           // [3,B,T,C]
    float* pred   = out + 1 + (size_t)NTASK * MROWS * CC;   // [3,B,T]

    unsigned char* bpws = (unsigned char*)d_ws;                      // 786432 B
    float* chain_loss = (float*)(bpws + (size_t)NCHAIN * BP_STRIDE); // 192 f32

    // 2048 M-tiles, each wave does both N-tiles; 8 waves/block -> 256 blocks
    scienceie_logits_wmma<<<256, 256, 0, stream>>>(enc, W, bias, logits);
    // 192 chains, 4 chains/wave -> 48 single-wave blocks
    scienceie_crf<<<NCHAIN / 4, 32, 0, stream>>>(logits, labels, start, endt,
                                                 trans, pred, bpws, chain_loss);
    scienceie_loss_reduce<<<1, 32, 0, stream>>>(chain_loss, loss);
}
